// Contextual_Reconstruction_Module_66159676228284
// MI455X (gfx1250) — matrix operand order repaired and checked by reading, not compiled, not run on hardware
//
#include <hip/hip_runtime.h>
#include <hip/hip_bf16.h>

// ---------------------------------------------------------------------------
// Contextual attention reconstruction for MI455X (gfx1250, wave32, WMMA).
//  k_prepw   : raw_w [L,C,4,4] f32  -> Wt [16][C][L] bf16 (A-operand layout)
//  k_fuse    : 9-point diagonal gather + mm*scale -> logits [L][4096] f32
//  k_softmax : softmax over L per spatial col, *mm -> yi [L][4096] bf16
//  k_deconv  : 4 parity planes x 4 taps of M64 x N4096 x K4096 bf16 GEMM
//              via v_wmma_f32_16x16x32_bf16; double-buffered LDS with
//              global_load_async_to_lds_b128 (ASYNCcnt) for the A operand
//  k_conv3x3 : two C64->C64 3x3 convs with ELU (direct VALU)
// ---------------------------------------------------------------------------

typedef __attribute__((ext_vector_type(16))) __bf16 v16bf;
typedef __attribute__((ext_vector_type(8)))  __bf16 v8bf;
typedef __attribute__((ext_vector_type(8)))  float  v8f;

#define NL   4096   // L (patches)
#define NSP  4096   // HS*WS (spatial)
#define NC   64

#define LDSA_ELEMS (4 * 64 * 40)   // [ta][c][k-padded-40] per buffer
#define LDSB_ELEMS (2 * 66 * 40)   // [row][t+1][k-padded-40] per buffer

// pass-2 diagonal shift on the transposed flattening (p +- d with carry into q)
__device__ __forceinline__ int shift2(int idx, int d) {
    int p = idx >> 6, q = idx & 63;
    int f = q * 64 + p + d;
    if (f < 0 || f >= 4096) return -1;
    return (f & 63) * 64 + (f >> 6);
}

// ------------------------------ fuse -----------------------------------
__global__ __launch_bounds__(256)
void k_fuse(const float* __restrict__ S, const float* __restrict__ mm,
            float* __restrict__ logits) {
    int c = blockIdx.x * 256 + threadIdx.x;   // spatial (contiguous)
    int r = blockIdx.y;                       // patch index
    float acc = 0.f;
#pragma unroll
    for (int d2 = -1; d2 <= 1; ++d2) {
        int rs2 = shift2(r, d2);
        int cs2 = shift2(c, d2);
        if (rs2 < 0 || cs2 < 0) continue;
#pragma unroll
        for (int d1 = -1; d1 <= 1; ++d1) {
            int rr = rs2 + d1, cc = cs2 + d1;
            if (rr >= 0 && rr < 4096 && cc >= 0 && cc < 4096)
                acc += S[rr * 4096 + cc];
        }
    }
    logits[r * 4096 + c] = acc * mm[r] * 10.0f;
}

// ----------------------------- softmax ---------------------------------
__global__ __launch_bounds__(256)
void k_softmax(const float* __restrict__ logits, const float* __restrict__ mm,
               __bf16* __restrict__ yi) {
    int c = blockIdx.x * 256 + threadIdx.x;   // one spatial column per thread
    float mx = -3.4e38f;
    for (int r = 0; r < NL; ++r) mx = fmaxf(mx, logits[r * 4096 + c]);
    float s = 0.f;
    for (int r = 0; r < NL; ++r) s += __expf(logits[r * 4096 + c] - mx);
    float inv = 1.0f / s;
    for (int r = 0; r < NL; ++r) {
        float v = __expf(logits[r * 4096 + c] - mx) * inv * mm[r];
        yi[r * 4096 + c] = (__bf16)v;
    }
}

// --------------------------- weight prep -------------------------------
// raw_w flat index = l*1024 + c*16 + tap  ->  Wt[(tap*64 + c)*4096 + l]
__global__ __launch_bounds__(256)
void k_prepw(const float* __restrict__ rw, __bf16* __restrict__ Wt) {
    int idx = blockIdx.x * 256 + threadIdx.x;   // 16*64*4096 total
    int l   = idx >> 10;
    int c   = (idx >> 4) & 63;
    int tap = idx & 15;
    Wt[((size_t)(tap * 64 + c) << 12) + l] = (__bf16)rw[idx];
}

// ---------------------- deconv staging / compute -----------------------
// stage one 32-K chunk: A (weights) via async-to-LDS copies, B (yi) via
// regular loads + ds stores (needs per-element zero padding at the halo).
__device__ __forceinline__ void stage_chunk(
    const __bf16* __restrict__ yi, const __bf16* __restrict__ Wt,
    __bf16* ldsAbuf, __bf16* ldsBbuf,
    int tid, int pu, int pv, int s_lo, int s_hi, int k0)
{
    // ---- A: 4 taps x 64 c x 32 k bf16, 64B contiguous per thread.
    // global_load_async_to_lds_b128: per-lane LDS byte addr in VGPR; the low
    // 32 bits of a flat shared pointer ARE the LDS offset (ISA 10.2).
    {
        int ta = tid >> 6;                 // 0..3 : ta = a*2+b
        int a  = ta >> 1, b = ta & 1;
        int tap = ((1 - pu) + 2 * a) * 4 + ((1 - pv) + 2 * b);
        int c   = tid & 63;
        uint64_t ga = (uint64_t)(uintptr_t)(Wt + (((size_t)(tap * 64 + c)) << 12) + k0);
        uint32_t la = (uint32_t)(uintptr_t)(void*)&ldsAbuf[(ta * 64 + c) * 40];
#pragma unroll
        for (int j = 0; j < 4; ++j) {
            asm volatile("global_load_async_to_lds_b128 %0, %1, off"
                         :: "v"(la + 16u * j), "v"(ga + 16ull * j)
                         : "memory");
        }
    }
    // ---- B: yi, 2 rows x 66 cols (t = -1..64, zero-padded) x 32 k
    for (int idx = tid; idx < 32 * 2 * 66; idx += 256) {
        int col = idx % 66;
        int rem = idx / 66;
        int row = rem & 1;
        int kk  = rem >> 1;
        int sA  = row ? s_hi : s_lo;
        int t   = col - 1;
        __bf16 v = (__bf16)0.0f;
        if (sA >= 0 && sA < 64 && t >= 0 && t < 64)
            v = yi[((size_t)(k0 + kk) << 12) + sA * 64 + t];
        ldsBbuf[(row * 66 + col) * 40 + kk] = v;
    }
}

__device__ __forceinline__ void compute_chunk(
    const __bf16* ldsAbuf, const __bf16* ldsBbuf,
    int wm, int wn, int mrow, int hi, int pv, v8f acc[2])
{
    // A fragments: lane row = c, K runs {kb..kb+7} and {kb+16..kb+23}
    v16bf afrag[4];
#pragma unroll
    for (int ta = 0; ta < 4; ++ta) {
        const int base = (ta * 64 + wm * 16 + mrow) * 40 + (hi ? 8 : 0);
        union { v16bf v; v8bf h[2]; } u;
        u.h[0] = *(const v8bf*)&ldsAbuf[base];
        u.h[1] = *(const v8bf*)&ldsAbuf[base + 16];
        afrag[ta] = u.v;
    }
    // 2 n-subtiles x 4 taps WMMA; B lane col = n, K runs {kb..kb+7} and {kb+16..kb+23}
#pragma unroll
    for (int sub = 0; sub < 2; ++sub) {
#pragma unroll
        for (int a = 0; a < 2; ++a) {
#pragma unroll
            for (int b = 0; b < 2; ++b) {
                const int ta  = a * 2 + b;
                const int tp  = wn * 32 + sub * 16 + mrow;   // t'
                const int col = tp + pv - b + 1;             // 0..65
                const int row = (a == 1) ? 0 : 1;
                const int base = (row * 66 + col) * 40 + hi * 8;
                union { v16bf v; v8bf h[2]; } u;
                u.h[0] = *(const v8bf*)&ldsBbuf[base];
                u.h[1] = *(const v8bf*)&ldsBbuf[base + 16];
                acc[sub] = __builtin_amdgcn_wmma_f32_16x16x32_bf16(
                    false, afrag[ta], false, u.v,
                    (short)0, acc[sub], false, false);
            }
        }
    }
}

// ------------------------- WMMA deconv GEMM ----------------------------
// block: 256 threads = 8 wave32s; grid: 4 parity planes * 64 s' rows.
// wave (wm in 0..3, wn in 0..1) computes c-tile wm*16..+16, t'-tile wn*32..+32.
// Double-buffered LDS: stage chunk i+1 (async A + ds B) while WMMA'ing chunk i.
__global__ __launch_bounds__(256)
void k_deconv(const __bf16* __restrict__ yi, const __bf16* __restrict__ Wt,
              float* __restrict__ y) {
    __shared__ __align__(16) __bf16 ldsA[2 * LDSA_ELEMS];
    __shared__ __align__(16) __bf16 ldsB[2 * LDSB_ELEMS];

    const int tid   = threadIdx.x;
    const int blk   = blockIdx.x;      // 0..255
    const int plane = blk >> 6;
    const int pu = plane >> 1, pv = plane & 1;
    const int srow  = blk & 63;        // s'

    const int s_lo = srow + pu - 1;    // lds row 0  (tap a = 1)
    const int s_hi = srow + pu;        // lds row 1  (tap a = 0)

    const int w    = tid >> 5;         // wave id (wave32)
    const int lane = tid & 31;
    const int wm = w & 3, wn = w >> 2;
    const int mrow = lane & 15;
    const int hi   = lane >> 4;

    v8f acc[2] = {v8f{}, v8f{}};

    // prologue: stage chunk 0 into buffer 0
    stage_chunk(yi, Wt, &ldsA[0], &ldsB[0], tid, pu, pv, s_lo, s_hi, 0);

    for (int i = 0; i < NL / 32; ++i) {
        const int cur = i & 1;
        // my async A copies (and everyone's ds stores, via the barrier) for
        // chunk i must have landed before anyone reads buffer `cur`.
        asm volatile("s_wait_asynccnt 0x0" ::: "memory");
        __syncthreads();
        if (i + 1 < NL / 32)
            stage_chunk(yi, Wt,
                        &ldsA[(1 - cur) * LDSA_ELEMS], &ldsB[(1 - cur) * LDSB_ELEMS],
                        tid, pu, pv, s_lo, s_hi, (i + 1) * 32);
        compute_chunk(&ldsA[cur * LDSA_ELEMS], &ldsB[cur * LDSB_ELEMS],
                      wm, wn, mrow, hi, pv, acc);
        // no trailing barrier needed: barrier i+1 can only complete after every
        // wave consumed buffer `cur` into VGPRs (s_wait_dscnt precedes its wmma).
    }

    // ---- epilogue: /4 overlap norm, scatter into interleaved [c][u][v]
    const int u_out = 2 * srow + pu;
#pragma unroll
    for (int sub = 0; sub < 2; ++sub) {
        const int v_out = 2 * (wn * 32 + sub * 16 + mrow) + pv;
#pragma unroll
        for (int r = 0; r < 8; ++r) {
            int cch = wm * 16 + hi * 8 + r;
            y[((size_t)cch * 128 + u_out) * 128 + v_out] = acc[sub][r] * 0.25f;
        }
    }
}

// ------------------------- 3x3 conv + ELU ------------------------------
__global__ __launch_bounds__(256)
void k_conv3x3_elu(const float* __restrict__ x, const float* __restrict__ wgt,
                   const float* __restrict__ bias, float* __restrict__ out) {
    int n  = blockIdx.x * 256 + threadIdx.x;   // 64*128*128
    int co = n >> 14;
    int p  = n & 16383;
    int uu = p >> 7, vv = p & 127;
    float acc = bias[co];
    for (int ci = 0; ci < 64; ++ci) {
        const float* xp = x + ci * 16384;
        const float* wp = wgt + (co * 64 + ci) * 9;   // scalar-broadcast reads
#pragma unroll
        for (int dy = 0; dy < 3; ++dy) {
            int yy = uu + dy - 1;
            if (yy < 0 || yy >= 128) continue;
#pragma unroll
            for (int dx = 0; dx < 3; ++dx) {
                int xx = vv + dx - 1;
                if (xx < 0 || xx >= 128) continue;
                acc = fmaf(xp[yy * 128 + xx], wp[dy * 3 + dx], acc);
            }
        }
    }
    out[n] = acc > 0.f ? acc : (__expf(acc) - 1.f);   // ELU(alpha=1)
}

// ------------------------------ launch ---------------------------------
extern "C" void kernel_launch(void* const* d_in, const int* in_sizes, int n_in,
                              void* d_out, int out_size, void* d_ws, size_t ws_size,
                              hipStream_t stream) {
    const float* raw_w  = (const float*)d_in[0];   // [1,4096,64,4,4]
    const float* mm     = (const float*)d_in[1];   // [1,4096,1,1]
    const float* scores = (const float*)d_in[2];   // [1,4096,64,64]
    const float* w1     = (const float*)d_in[3];
    const float* b1     = (const float*)d_in[4];
    const float* w2     = (const float*)d_in[5];
    const float* b2     = (const float*)d_in[6];
    float* out = (float*)d_out;                    // [1,64,128,128]

    char* ws = (char*)d_ws;
    float*  logits = (float*)(ws);                         // 64 MB
    __bf16* yi     = (__bf16*)(ws + (size_t)67108864);     // 32 MB
    __bf16* Wt     = (__bf16*)(ws + (size_t)100663296);    //  8 MB
    float*  y0     = (float*)(ws + (size_t)109051904);     //  4 MB
    float*  y1     = (float*)(ws + (size_t)113246208);     //  4 MB

    k_prepw<<<16384, 256, 0, stream>>>(raw_w, Wt);
    dim3 gf(16, 4096);
    k_fuse<<<gf, 256, 0, stream>>>(scores, mm, logits);
    k_softmax<<<16, 256, 0, stream>>>(logits, mm, yi);
    k_deconv<<<256, 256, 0, stream>>>(yi, Wt, y0);
    k_conv3x3_elu<<<4096, 256, 0, stream>>>(y0, w1, b1, y1);
    k_conv3x3_elu<<<4096, 256, 0, stream>>>(y1, w2, b2, out);
}
